// NSMCell_6227702579421
// MI455X (gfx1250) — compile-verified, matrix operand order checked by reading, not run on hardware
//
#include <hip/hip_runtime.h>
#include <hip/hip_bf16.h>
#include <math.h>

typedef __attribute__((ext_vector_type(16))) _Float16 v16h;
typedef __attribute__((ext_vector_type(8)))  float    v8f;

// ---------------- helpers ----------------

__device__ __forceinline__ unsigned fkey(float f) {
    unsigned b = __float_as_uint(f);
    return (b & 0x80000000u) ? ~b : (b | 0x80000000u);
}
__device__ __forceinline__ float funkey(unsigned k) {
    unsigned b = (k & 0x80000000u) ? (k & 0x7FFFFFFFu) : ~k;
    return __uint_as_float(b);
}
__device__ __forceinline__ float elu1(float x) {
    return x > 0.0f ? x : (expf(x) - 1.0f);
}

// Unconditional: 8 consecutive fp32 -> f16 into frag[off..off+7].
// ptr is 16B aligned (row starts are multiples of 4 floats, chunk offsets multiples of 4).
__device__ __forceinline__ void load8_fast(const float* __restrict__ p, v16h& frag, int off) {
    const float4 a = *(const float4*)(p);
    const float4 b = *(const float4*)(p + 4);
    frag[off + 0] = (_Float16)a.x; frag[off + 1] = (_Float16)a.y;
    frag[off + 2] = (_Float16)a.z; frag[off + 3] = (_Float16)a.w;
    frag[off + 4] = (_Float16)b.x; frag[off + 5] = (_Float16)b.y;
    frag[off + 6] = (_Float16)b.z; frag[off + 7] = (_Float16)b.w;
}

// Branchless tail: clamped address + select-to-zero (no exec-mask branches).
__device__ __forceinline__ void load8_tail(const float* __restrict__ base, int kstart, int kmax,
                                           v16h& frag, int off) {
    #pragma unroll
    for (int i = 0; i < 8; ++i) {
        int k  = kstart + i;
        int kc = k < kmax ? k : (kmax - 1);
        float v = base[kc];
        v = (k < kmax) ? v : 0.0f;
        frag[off + i] = (_Float16)v;
    }
}

// ---------------- kernel 1: prop_sim = softmax(instruction @ prop_embeds^T) ----------------
__global__ void prop_sim_kernel(const float* __restrict__ instruction,
                                const float* __restrict__ prop_embeds,
                                float* __restrict__ ws_prop,   // [B,8] padded
                                float* __restrict__ out_tail,  // [B,P]
                                int H, int P) {
    const int b = blockIdx.x;
    const int lane = threadIdx.x;
    float acc[8];
    #pragma unroll
    for (int j = 0; j < 8; ++j) acc[j] = 0.0f;
    for (int k = lane; k < H; k += 32) {
        float x = instruction[(size_t)b * H + k];
        for (int j = 0; j < P; ++j) acc[j] += x * prop_embeds[(size_t)j * H + k];
    }
    for (int j = 0; j < P; ++j)
        for (int m = 16; m >= 1; m >>= 1) acc[j] += __shfl_xor(acc[j], m, 32);
    if (lane == 0) {
        float mx = acc[0];
        for (int j = 1; j < P; ++j) mx = fmaxf(mx, acc[j]);
        float e[8], s = 0.0f;
        for (int j = 0; j < P; ++j) { e[j] = expf(acc[j] - mx); s += e[j]; }
        float inv = 1.0f / s;
        for (int j = 0; j < P; ++j) {
            float v = e[j] * inv;
            ws_prop[b * 8 + j]  = v;
            out_tail[b * P + j] = v;
        }
    }
}

// ---------------- kernel 0: init scratch ----------------
__global__ void init_kernel(float* state_logits, float* rel_logits,
                            unsigned* segmax_s, unsigned* segmax_r,
                            float* segsum_s, float* segsum_r, int N, int B) {
    int i = blockIdx.x * blockDim.x + threadIdx.x;
    int stride = gridDim.x * blockDim.x;
    for (int n = i; n < N; n += stride) { state_logits[n] = 0.0f; rel_logits[n] = 0.0f; }
    for (int b = i; b < B; b += stride) {
        segmax_s[b] = 0u; segmax_r[b] = 0u; segsum_s[b] = 0.0f; segsum_r[b] = 0.0f;
    }
}

// ---------------- kernel 2: node GEMM (f16 WMMA, fp32 accum) + fused epilogue ----------------
// grid: (ceil(H/16), ceil(N/128)); block: 256 (8 waves); wave w -> 16 rows x one 16-col tile.
__global__ __launch_bounds__(256)
void node_gemm_kernel(const float* __restrict__ node_attrs,
                      const float* __restrict__ Ws,          // Ws_property[0..3] base
                      const float* __restrict__ instruction,
                      const int*   __restrict__ node_indices,
                      const float* __restrict__ ws_prop,     // [B,8]
                      const float* __restrict__ W_state,
                      float* __restrict__ state_logits,
                      int N, int H) {
    const int lane  = threadIdx.x & 31;
    const int wave  = threadIdx.x >> 5;
    const int hbase = blockIdx.x * 16;
    const int mbase = blockIdx.y * 128 + wave * 16;
    const int hi  = lane >> 4;       // half-wave select
    const int lo  = lane & 15;
    const int klo = hi * 8;          // ISA 16-bit A/B layout K offset per half-wave

    v8f acc0 = (v8f)0.0f, acc1 = (v8f)0.0f, acc2 = (v8f)0.0f, acc3 = (v8f)0.0f;

    int rowA = mbase + lo; if (rowA >= N) rowA = N - 1;          // clamp; padded rows discarded
    const float* Abase = node_attrs + (size_t)rowA * H + klo;
    int hB = hbase + lo; if (hB >= H) hB = H - 1;                // clamp; padded cols of D live
    const size_t pstride = (size_t)H * H;                        //   only in discarded lanes
    const float* Bb0 = Ws + (size_t)hB * H + klo;

    int kb = 0;
    for (; kb + 32 <= H; kb += 32) {                             // fully in-range: branch-free
        v16h a;
        load8_fast(Abase + kb,      a, 0);
        load8_fast(Abase + kb + 16, a, 8);
        #pragma unroll
        for (int p = 0; p < 4; ++p) {
            v16h bf;
            const float* Bb = Bb0 + p * pstride + kb;
            load8_fast(Bb,      bf, 0);
            load8_fast(Bb + 16, bf, 8);
            v8f& acc = (p == 0) ? acc0 : (p == 1) ? acc1 : (p == 2) ? acc2 : acc3;
            acc = __builtin_amdgcn_wmma_f32_16x16x32_f16(false, a, false, bf,
                                                         (short)0, acc, false, false);
        }
    }
    if (kb < H) {                                                // single tail step, branchless loads
        const float* Ab = node_attrs + (size_t)rowA * H;
        v16h a;
        load8_tail(Ab, kb + klo,      H, a, 0);
        load8_tail(Ab, kb + klo + 16, H, a, 8);
        #pragma unroll
        for (int p = 0; p < 4; ++p) {
            v16h bf;
            const float* Bb = Ws + (size_t)p * pstride + (size_t)hB * H;
            load8_tail(Bb, kb + klo,      H, bf, 0);
            load8_tail(Bb, kb + klo + 16, H, bf, 8);
            v8f& acc = (p == 0) ? acc0 : (p == 1) ? acc1 : (p == 2) ? acc2 : acc3;
            acc = __builtin_amdgcn_wmma_f32_16x16x32_f16(false, a, false, bf,
                                                         (short)0, acc, false, false);
        }
    }

    // Epilogue: score = elu(instr[idx,col] * sum_p ps[idx,p]*acc_p); contribute score*W_state[col]
    const int col = hbase + lo;
    const float wst = (col < H) ? W_state[col] : 0.0f;
    #pragma unroll
    for (int r = 0; r < 8; ++r) {
        const int row = mbase + r + 8 * hi;
        float contrib = 0.0f;
        if (row < N && col < H) {
            const int idx = node_indices[row];
            const float4 ps = *(const float4*)(ws_prop + idx * 8);
            float s = ps.x * acc0[r] + ps.y * acc1[r] + ps.z * acc2[r] + ps.w * acc3[r];
            float v = instruction[(size_t)idx * H + col] * s;
            contrib = elu1(v) * wst;
        }
        for (int m = 8; m >= 1; m >>= 1) contrib += __shfl_xor(contrib, m, 32);
        if (lo == 0 && row < N) atomicAdd(&state_logits[row], contrib);
    }
}

// ---------------- kernel 3: edge GEMM (p=4) + fused scatter epilogue ----------------
__global__ __launch_bounds__(256)
void edge_gemm_kernel(const float* __restrict__ edge_attrs,
                      const float* __restrict__ Ws_last,     // Ws_property[P-1]
                      const float* __restrict__ instruction,
                      const int*   __restrict__ edge_batch,
                      const int*   __restrict__ edge_indices, // [2,E] flat
                      const float* __restrict__ distribution,
                      const float* __restrict__ W_relation,
                      float* __restrict__ rel_logits,
                      int E, int H) {
    const int lane  = threadIdx.x & 31;
    const int wave  = threadIdx.x >> 5;
    const int hbase = blockIdx.x * 16;
    const int ebase = blockIdx.y * 128 + wave * 16;
    const int hi  = lane >> 4;
    const int lo  = lane & 15;
    const int klo = hi * 8;

    v8f acc = (v8f)0.0f;

    int rowA = ebase + lo; if (rowA >= E) rowA = E - 1;
    const float* Abase = edge_attrs + (size_t)rowA * H + klo;
    int hB = hbase + lo; if (hB >= H) hB = H - 1;
    const float* Bb0 = Ws_last + (size_t)hB * H + klo;

    int kb = 0;
    for (; kb + 32 <= H; kb += 32) {
        v16h a, bf;
        load8_fast(Abase + kb,      a, 0);
        load8_fast(Abase + kb + 16, a, 8);
        load8_fast(Bb0 + kb,        bf, 0);
        load8_fast(Bb0 + kb + 16,   bf, 8);
        acc = __builtin_amdgcn_wmma_f32_16x16x32_f16(false, a, false, bf,
                                                     (short)0, acc, false, false);
    }
    if (kb < H) {
        const float* Ab = edge_attrs + (size_t)rowA * H;
        const float* Bb = Ws_last + (size_t)hB * H;
        v16h a, bf;
        load8_tail(Ab, kb + klo,      H, a, 0);
        load8_tail(Ab, kb + klo + 16, H, a, 8);
        load8_tail(Bb, kb + klo,      H, bf, 0);
        load8_tail(Bb, kb + klo + 16, H, bf, 8);
        acc = __builtin_amdgcn_wmma_f32_16x16x32_f16(false, a, false, bf,
                                                     (short)0, acc, false, false);
    }

    const int col = hbase + lo;
    const float wrl = (col < H) ? W_relation[col] : 0.0f;
    #pragma unroll
    for (int r = 0; r < 8; ++r) {
        const int e = ebase + r + 8 * hi;
        float contrib = 0.0f;
        if (e < E && col < H) {
            const int b = edge_batch[e];
            float v = instruction[(size_t)b * H + col] * acc[r];
            contrib = elu1(v) * wrl;
        }
        for (int m = 8; m >= 1; m >>= 1) contrib += __shfl_xor(contrib, m, 32);
        if (lo == 0 && e < E) {
            const int src = edge_indices[e];
            const int dst = edge_indices[E + e];
            atomicAdd(&rel_logits[dst], distribution[src] * contrib);
        }
    }
}

// ---------------- segment softmax (3 passes) ----------------
__global__ void seg_max_kernel(const float* __restrict__ sl, const float* __restrict__ rl,
                               const int* __restrict__ seg,
                               unsigned* __restrict__ ms, unsigned* __restrict__ mr, int N) {
    int i = blockIdx.x * blockDim.x + threadIdx.x;
    if (i < N) {
        int s = seg[i];
        atomicMax(&ms[s], fkey(sl[i]));
        atomicMax(&mr[s], fkey(rl[i]));
    }
}

__global__ void seg_expsum_kernel(const float* __restrict__ sl, const float* __restrict__ rl,
                                  const int* __restrict__ seg,
                                  const unsigned* __restrict__ ms, const unsigned* __restrict__ mr,
                                  float* __restrict__ es, float* __restrict__ er,
                                  float* __restrict__ ss, float* __restrict__ sr, int N) {
    int i = blockIdx.x * blockDim.x + threadIdx.x;
    if (i < N) {
        int s = seg[i];
        float e1 = expf(sl[i] - funkey(ms[s])); es[i] = e1; atomicAdd(&ss[s], e1);
        float e2 = expf(rl[i] - funkey(mr[s])); er[i] = e2; atomicAdd(&sr[s], e2);
    }
}

__global__ void finalize_kernel(const int* __restrict__ seg,
                                const float* __restrict__ es, const float* __restrict__ er,
                                const float* __restrict__ ss, const float* __restrict__ sr,
                                const float* __restrict__ ws_prop,
                                float* __restrict__ out, int N, int P) {
    int i = blockIdx.x * blockDim.x + threadIdx.x;
    if (i < N) {
        int s = seg[i];
        float ns = es[i] / ss[s];
        float nr = er[i] / sr[s];
        float r  = ws_prop[s * 8 + (P - 1)];
        out[i] = r * nr + (1.0f - r) * ns;
    }
}

// ---------------- launcher ----------------
extern "C" void kernel_launch(void* const* d_in, const int* in_sizes, int n_in,
                              void* d_out, int out_size, void* d_ws, size_t ws_size,
                              hipStream_t stream) {
    const float* instruction  = (const float*)d_in[0];
    const float* distribution = (const float*)d_in[1];
    const float* node_attrs   = (const float*)d_in[2];
    const float* edge_attrs   = (const float*)d_in[3];
    const int*   node_indices = (const int*)  d_in[4];
    const int*   edge_batch   = (const int*)  d_in[5];
    const int*   edge_indices = (const int*)  d_in[6];
    const float* prop_embeds  = (const float*)d_in[7];
    const float* Ws_property  = (const float*)d_in[8];
    const float* W_state      = (const float*)d_in[9];
    const float* W_relation   = (const float*)d_in[10];

    const int H = in_sizes[9];          // W_state
    const int B = in_sizes[0] / H;      // instruction B*H
    const int N = in_sizes[1];          // distribution
    const int E = in_sizes[3] / H;      // edge_attrs E*H
    const int P = in_sizes[7] / H;      // prop_embeds P*H

    float* out = (float*)d_out;
    float* out_tail = out + N;          // prop_sim lives after next_distribution

    auto alignup = [](size_t x) { return (x + 255) & ~(size_t)255; };
    char* w = (char*)d_ws;
    size_t off = 0;
    float*    ws_prop      = (float*)(w + off); off += alignup((size_t)B * 8 * sizeof(float));
    float*    state_logits = (float*)(w + off); off += alignup((size_t)N * sizeof(float));
    float*    rel_logits   = (float*)(w + off); off += alignup((size_t)N * sizeof(float));
    float*    e_state      = (float*)(w + off); off += alignup((size_t)N * sizeof(float));
    float*    e_rel        = (float*)(w + off); off += alignup((size_t)N * sizeof(float));
    unsigned* segmax_s     = (unsigned*)(w + off); off += alignup((size_t)B * sizeof(unsigned));
    unsigned* segmax_r     = (unsigned*)(w + off); off += alignup((size_t)B * sizeof(unsigned));
    float*    segsum_s     = (float*)(w + off); off += alignup((size_t)B * sizeof(float));
    float*    segsum_r     = (float*)(w + off); off += alignup((size_t)B * sizeof(float));
    (void)ws_size; (void)n_in; (void)out_size;

    const int nblk = (N + 255) / 256;

    init_kernel<<<nblk, 256, 0, stream>>>(state_logits, rel_logits,
                                          segmax_s, segmax_r, segsum_s, segsum_r, N, B);

    prop_sim_kernel<<<B, 32, 0, stream>>>(instruction, prop_embeds, ws_prop, out_tail, H, P);

    {
        dim3 grid((H + 15) / 16, (N + 127) / 128);
        node_gemm_kernel<<<grid, 256, 0, stream>>>(node_attrs, Ws_property, instruction,
                                                   node_indices, ws_prop, W_state,
                                                   state_logits, N, H);
    }
    {
        dim3 grid((H + 15) / 16, (E + 127) / 128);
        const float* Ws_last = Ws_property + (size_t)(P - 1) * H * H;
        edge_gemm_kernel<<<grid, 256, 0, stream>>>(edge_attrs, Ws_last, instruction,
                                                   edge_batch, edge_indices, distribution,
                                                   W_relation, rel_logits, E, H);
    }

    seg_max_kernel<<<nblk, 256, 0, stream>>>(state_logits, rel_logits, node_indices,
                                             segmax_s, segmax_r, N);
    seg_expsum_kernel<<<nblk, 256, 0, stream>>>(state_logits, rel_logits, node_indices,
                                                segmax_s, segmax_r,
                                                e_state, e_rel, segsum_s, segsum_r, N);
    finalize_kernel<<<nblk, 256, 0, stream>>>(node_indices, e_state, e_rel,
                                              segsum_s, segsum_r, ws_prop, out, N, P);
}
